// AtrousDualAttention_50448685858977
// MI455X (gfx1250) — matrix operand order repaired and checked by reading, not compiled, not run on hardware
//
#include <hip/hip_runtime.h>
#include <hip/hip_bf16.h>
#include <math.h>

typedef __attribute__((ext_vector_type(16))) _Float16 v16h;
typedef __attribute__((ext_vector_type(8)))  _Float16 v8h;
typedef __attribute__((ext_vector_type(8)))  float    v8f;

#define CB    2
#define CC    256
#define CH    80
#define CW    80
#define CHW   6400          // 80*80
#define NALL  12800         // B*HW
#define CI    16            // INTER
#define K5    2304          // 9*256
#define K3P   160           // 9*16 padded to 160

// ---------------- WMMA helpers ----------------
__device__ inline v8f wmma16(v16h a, v16h b, v8f c) {
  return __builtin_amdgcn_wmma_f32_16x16x32_f16(false, a, false, b, (short)0, c, false, false);
}

// A operand: row-major f16 matrix, row m = lane&15.
// A layout (contiguous, matches B): lane half h: elem j -> K=k0+h*16+j
__device__ inline v16h load_a16(const _Float16* A, int lda, int k0, int lane) {
  int m = lane & 15, h = lane >> 4;
  const _Float16* p = A + m * lda + k0 + h * 16;
  v8h lo = *(const v8h*)(p);
  v8h hi = *(const v8h*)(p + 8);
  return __builtin_shufflevector(lo, hi, 0,1,2,3,4,5,6,7,8,9,10,11,12,13,14,15);
}

// ---------------- 1. depthwise dilated convs ----------------
__global__ void k_depthwise(const float* __restrict__ z,
                            const float* w0, const float* b0, const float* w1, const float* b1,
                            const float* w2, const float* b2, const float* w3, const float* b3,
                            float* __restrict__ zc) {
  int idx = blockIdx.x * 256 + threadIdx.x;
  if (idx >= CB * CC * CHW) return;
  int n = idx % CHW; int c = (idx / CHW) % CC; int b = idx / (CHW * CC);
  int y = n / CW, x = n % CW;
  int qd = c >> 6, cl = c & 63;
  const int dils[4] = {7, 5, 2, 1};
  const float* wq = qd == 0 ? w0 : qd == 1 ? w1 : qd == 2 ? w2 : w3;
  const float* bq = qd == 0 ? b0 : qd == 1 ? b1 : qd == 2 ? b2 : b3;
  int d = dils[qd];
  float acc = bq[cl];
  const float* zi = z + (size_t)(b * CC + c) * CHW;
  #pragma unroll
  for (int ky = 0; ky < 3; ++ky)
    #pragma unroll
    for (int kx = 0; kx < 3; ++kx) {
      int yy = y + (ky - 1) * d, xx = x + (kx - 1) * d;
      if ((unsigned)yy < (unsigned)CH && (unsigned)xx < (unsigned)CW)
        acc += wq[cl * 9 + ky * 3 + kx] * zi[yy * CW + xx];
    }
  zc[idx] = acc;
}

// ---------------- 2. GroupNorm stats (4 groups of 64 ch) ----------------
__global__ void k_gnstats(const float* __restrict__ zc, float* gstats) {
  int bg = blockIdx.x;          // b*4+g
  int b = bg >> 2, g = bg & 3;
  const float* p = zc + (size_t)(b * CC + g * 64) * CHW;
  float s = 0.f, s2 = 0.f;
  for (int i = threadIdx.x; i < 64 * CHW; i += 256) { float v = p[i]; s += v; s2 += v * v; }
  __shared__ float ls[256], ls2[256];
  ls[threadIdx.x] = s; ls2[threadIdx.x] = s2; __syncthreads();
  for (int st = 128; st > 0; st >>= 1) {
    if (threadIdx.x < st) { ls[threadIdx.x] += ls[threadIdx.x + st]; ls2[threadIdx.x] += ls2[threadIdx.x + st]; }
    __syncthreads();
  }
  if (threadIdx.x == 0) {
    float inv = 1.0f / (64.0f * CHW);
    float m = ls[0] * inv, var = ls2[0] * inv - m * m;
    gstats[bg * 2] = m; gstats[bg * 2 + 1] = rsqrtf(var + 1e-5f);
  }
}

// ---------------- 3. GroupNorm apply -> transposed f16 [pos][256] ----------------
__global__ void k_gnapply(const float* __restrict__ zc, const float* gstats,
                          const float* gw, const float* gb, _Float16* __restrict__ xT) {
  int idx = blockIdx.x * 256 + threadIdx.x;
  if (idx >= CB * CC * CHW) return;
  int n = idx % CHW; int c = (idx / CHW) % CC; int b = idx / (CHW * CC);
  int bg = b * 4 + (c >> 6);
  float m = gstats[bg * 2], rs = gstats[bg * 2 + 1];
  float v = (zc[idx] - m) * rs * gw[c] + gb[c];
  xT[(size_t)(b * CHW + n) * CC + c] = (_Float16)v;
}

// ---------------- 4. weight prep (all f16 repacks) ----------------
__global__ void k_prep_weights(const float* conv_w, const float* c5a, const float* c5c,
                               const float* c5b, const float* c5d, const float* c6,
                               _Float16* W1h, _Float16* W2a, _Float16* W2c,
                               _Float16* W3b, _Float16* W3d, _Float16* W6) {
  int i = blockIdx.x * 256 + threadIdx.x;
  if (i < 65536) { W1h[i] = (_Float16)conv_w[i]; return; } i -= 65536;
  if (i < 36864) { int oc = i / K5, k = i % K5, tap = k >> 8, ic = k & 255;
                   W2a[i] = (_Float16)c5a[(oc * 256 + ic) * 9 + tap]; return; } i -= 36864;
  if (i < 36864) { int oc = i / K5, k = i % K5, tap = k >> 8, ic = k & 255;
                   W2c[i] = (_Float16)c5c[(oc * 256 + ic) * 9 + tap]; return; } i -= 36864;
  if (i < 2560)  { int oc = i / K3P, k = i % K3P; _Float16 v = (_Float16)0.f;
                   if (k < 144) { int tap = k >> 4, ic = k & 15; v = (_Float16)c5b[(oc * 16 + ic) * 9 + tap]; }
                   W3b[i] = v; return; } i -= 2560;
  if (i < 2560)  { int oc = i / K3P, k = i % K3P; _Float16 v = (_Float16)0.f;
                   if (k < 144) { int tap = k >> 4, ic = k & 15; v = (_Float16)c5d[(oc * 16 + ic) * 9 + tap]; }
                   W3d[i] = v; return; } i -= 2560;
  if (i < 8192)  { int oc = i / 32, k = i % 32, ic = k & 15; W6[i] = (_Float16)c6[oc * 16 + ic]; }
}

// ---------------- 5. 1x1 conv 256->256 + bias + exact GELU (WMMA) ----------------
__global__ void k_gemm_pw1(const _Float16* __restrict__ xT, const _Float16* __restrict__ W1h,
                           const float* __restrict__ bias, _Float16* __restrict__ zc2T) {
  int lane = threadIdx.x, t = blockIdx.x, oc0 = blockIdx.y * 16;
  int col = lane & 15, h = lane >> 4;
  int ng = t * 16 + col;
  const _Float16* brow = xT + (size_t)ng * CC;
  v8f acc = {};
  #pragma unroll
  for (int k0 = 0; k0 < CC; k0 += 32) {
    v16h a = load_a16(W1h + oc0 * CC, CC, k0, lane);
    v16h bv = *(const v16h*)(brow + k0 + h * 16);
    acc = wmma16(a, bv, acc);
  }
  #pragma unroll
  for (int r = 0; r < 8; ++r) {
    int oc = oc0 + h * 8 + r;
    float v = acc[r] + bias[oc];
    v = 0.5f * v * (1.0f + erff(v * 0.70710678118f));       // exact GELU
    zc2T[(size_t)ng * CC + oc] = (_Float16)v;
  }
}

// ---------------- 6. 3x3 conv 256->16, implicit im2col (WMMA), K=2304 ----------------
__global__ void k_gemm_c5(const _Float16* __restrict__ W2, const _Float16* __restrict__ zin,
                          float* __restrict__ out) {
  int lane = threadIdx.x;
  int t = blockIdx.x * 4 + threadIdx.y;
  int col = lane & 15, h = lane >> 4;
  int ng = t * 16 + col, b = ng / CHW, n = ng % CHW;
  int y = n / CW, x = n % CW;
  v8f acc = {};
  for (int k0 = 0; k0 < K5; k0 += 32) {
    v16h a = load_a16(W2, K5, k0, lane);
    int kk = k0 + h * 16, tap = kk >> 8, icb = kk & 255;
    int yy = y + tap / 3 - 1, xx = x + tap % 3 - 1;
    v16h bv = {};
    if ((unsigned)yy < (unsigned)CH && (unsigned)xx < (unsigned)CW)
      bv = *(const v16h*)(zin + ((size_t)(b * CHW + yy * CW + xx) * CC + icb));
    acc = wmma16(a, bv, acc);
  }
  #pragma unroll
  for (int r = 0; r < 8; ++r) { int c = h * 8 + r; out[(size_t)(b * CI + c) * CHW + n] = acc[r]; }
}

// ---------------- 7. BatchNorm stats (16 ch over B*HW), eps=1e-3 ----------------
__global__ void k_bnstats16(const float* __restrict__ xin, float* st) {
  int c = blockIdx.x;
  float s = 0.f, s2 = 0.f;
  for (int i = threadIdx.x; i < NALL; i += 256) {
    int b = i / CHW, n = i % CHW;
    float v = xin[(size_t)(b * CI + c) * CHW + n]; s += v; s2 += v * v;
  }
  __shared__ float ls[256], ls2[256];
  ls[threadIdx.x] = s; ls2[threadIdx.x] = s2; __syncthreads();
  for (int stp = 128; stp > 0; stp >>= 1) {
    if (threadIdx.x < stp) { ls[threadIdx.x] += ls[threadIdx.x + stp]; ls2[threadIdx.x] += ls2[threadIdx.x + stp]; }
    __syncthreads();
  }
  if (threadIdx.x == 0) {
    float m = ls[0] / (float)NALL, var = ls2[0] / (float)NALL - m * m;
    st[c * 2] = m; st[c * 2 + 1] = rsqrtf(var + 1e-3f);
  }
}

// ---------------- 8. BN apply + relu -> f16 [b][16][HW] ----------------
__global__ void k_bnapply16(const float* __restrict__ xin, const float* st,
                            const float* g, const float* bb, _Float16* __restrict__ oh) {
  int idx = blockIdx.x * 256 + threadIdx.x;
  if (idx >= CB * CI * CHW) return;
  int c = (idx / CHW) % CI;
  float v = (xin[idx] - st[c * 2]) * st[c * 2 + 1] * g[c] + bb[c];
  oh[idx] = (_Float16)fmaxf(v, 0.f);
}

// ---------------- 9. q,k,v 1x1 projections from feat1 ----------------
__global__ void k_qkv(const _Float16* __restrict__ f1,
                      const float* pqw, const float* pqb, const float* pkw, const float* pkb,
                      const float* pvw, const float* pvb,
                      float* __restrict__ q, float* __restrict__ kk, _Float16* __restrict__ vh) {
  int i = blockIdx.x * 256 + threadIdx.x;
  if (i >= NALL) return;
  int b = i / CHW, n = i % CHW;
  float f[16];
  #pragma unroll
  for (int c = 0; c < 16; ++c) f[c] = (float)f1[(size_t)(b * CI + c) * CHW + n];
  #pragma unroll
  for (int o = 0; o < 2; ++o) {
    float sq = pqb[o], sk = pkb[o];
    #pragma unroll
    for (int c = 0; c < 16; ++c) { sq += pqw[o * 16 + c] * f[c]; sk += pkw[o * 16 + c] * f[c]; }
    q[(size_t)(b * 2 + o) * CHW + n] = sq;
    kk[(size_t)(b * 2 + o) * CHW + n] = sk;
  }
  #pragma unroll
  for (int o = 0; o < 16; ++o) {
    float sv = pvb[o];
    #pragma unroll
    for (int c = 0; c < 16; ++c) sv += pvw[o * 16 + c] * f[c];
    vh[(size_t)(b * CI + o) * CHW + n] = (_Float16)sv;
  }
}

// ---------------- 10. PAM softmax row stats (online max/sum) ----------------
__global__ void k_pam_rowstats(const float* __restrict__ q, const float* __restrict__ kk,
                               float* __restrict__ rmax, float* __restrict__ rsum) {
  int i = blockIdx.x * 256 + threadIdx.x;
  if (i >= NALL) return;
  int b = i / CHW, n = i % CHW;
  float q0 = q[(size_t)(b * 2) * CHW + n], q1 = q[(size_t)(b * 2 + 1) * CHW + n];
  const float* k0p = kk + (size_t)(b * 2) * CHW;
  const float* k1p = k0p + CHW;
  float mx = -1e30f, sm = 0.f;
  for (int m = 0; m < CHW; ++m) {
    float s = q0 * k0p[m] + q1 * k1p[m];
    if (s > mx) { sm = sm * __expf(mx - s) + 1.f; mx = s; }
    else        { sm += __expf(s - mx); }
  }
  rmax[i] = mx; rsum[i] = sm;
}

// ---------------- 11. PAM PV product (flash pass 2, WMMA) ----------------
__global__ void k_pam_pv(const float* __restrict__ q, const float* __restrict__ kk,
                         const float* __restrict__ rmax, const float* __restrict__ rsum,
                         const _Float16* __restrict__ vh, const _Float16* __restrict__ f1,
                         const float* gamma, _Float16* __restrict__ paT) {
  int lane = threadIdx.x;
  int t = blockIdx.x * 4 + threadIdx.y;
  int col = lane & 15, h = lane >> 4;
  int ng = t * 16 + col, b = ng / CHW, n = ng % CHW;
  float q0 = q[(size_t)(b * 2) * CHW + n], q1 = q[(size_t)(b * 2 + 1) * CHW + n];
  float mx = rmax[ng], inv = 1.0f / rsum[ng];
  const float* k0p = kk + (size_t)(b * 2) * CHW;
  const float* k1p = k0p + CHW;
  const _Float16* vb = vh + (size_t)b * CI * CHW;
  v8f acc = {};
  for (int m0 = 0; m0 < CHW; m0 += 32) {
    v16h a = load_a16(vb, CHW, m0, lane);
    v16h bv = {};
    #pragma unroll
    for (int j = 0; j < 16; ++j) {
      int m = m0 + h * 16 + j;
      float s = q0 * k0p[m] + q1 * k1p[m];
      bv[j] = (_Float16)__expf(s - mx);
    }
    acc = wmma16(a, bv, acc);
  }
  float g = gamma[0];
  #pragma unroll
  for (int r = 0; r < 8; ++r) {
    int c = h * 8 + r;
    float val = g * acc[r] * inv + (float)f1[(size_t)(b * CI + c) * CHW + n];
    paT[(size_t)ng * CI + c] = (_Float16)val;
  }
}

// ---------------- 12. CAM gram e = x x^T (16x16, K=6400) ----------------
__global__ void k_cam_gram(const _Float16* __restrict__ f2, float* __restrict__ e) {
  int lane = threadIdx.x, b = blockIdx.x;
  int d = lane & 15, h = lane >> 4;
  const _Float16* base = f2 + (size_t)b * CI * CHW;
  v8f acc = {};
  for (int k0 = 0; k0 < CHW; k0 += 32) {
    v16h a = load_a16(base, CHW, k0, lane);
    v16h bv = *(const v16h*)(base + (size_t)d * CHW + k0 + h * 16);
    acc = wmma16(a, bv, acc);
  }
  #pragma unroll
  for (int r = 0; r < 8; ++r) { int c = h * 8 + r; e[b * 256 + c * 16 + d] = acc[r]; }
}

// ---------------- 13. CAM softmax: att = softmax(rowmax(e)-e) ----------------
__global__ void k_cam_softmax(const float* __restrict__ e, float* __restrict__ att) {
  int c = threadIdx.x, b = blockIdx.x;
  if (c >= 16) return;
  float row[16], mxE = -1e30f, mn = 1e30f;
  #pragma unroll
  for (int d = 0; d < 16; ++d) { row[d] = e[b * 256 + c * 16 + d]; mxE = fmaxf(mxE, row[d]); mn = fminf(mn, row[d]); }
  float m2 = mxE - mn, sm = 0.f, p[16];
  #pragma unroll
  for (int d = 0; d < 16; ++d) { p[d] = __expf((mxE - row[d]) - m2); sm += p[d]; }
  #pragma unroll
  for (int d = 0; d < 16; ++d) att[b * 256 + c * 16 + d] = p[d] / sm;
}

// ---------------- 14. CAM apply: ca = gamma*(att@x)+x -> f16 [pos][16] ----------------
__global__ void k_cam_apply(const _Float16* __restrict__ f2, const float* __restrict__ att,
                            const float* gamma, _Float16* __restrict__ caT) {
  int i = blockIdx.x * 256 + threadIdx.x;
  if (i >= NALL) return;
  int b = i / CHW, n = i % CHW;
  float f[16];
  #pragma unroll
  for (int d = 0; d < 16; ++d) f[d] = (float)f2[(size_t)(b * CI + d) * CHW + n];
  float g = gamma[0];
  #pragma unroll
  for (int c = 0; c < 16; ++c) {
    float s = 0.f;
    #pragma unroll
    for (int d = 0; d < 16; ++d) s += att[b * 256 + c * 16 + d] * f[d];
    caT[(size_t)i * CI + c] = (_Float16)(g * s + f[c]);
  }
}

// ---------------- 15. 3x3 conv 16->16, K padded to 160 (WMMA) ----------------
__global__ void k_gemm_small(const _Float16* __restrict__ W3, const _Float16* __restrict__ inT,
                             float* __restrict__ out) {
  int lane = threadIdx.x;
  int t = blockIdx.x * 4 + threadIdx.y;
  int col = lane & 15, h = lane >> 4;
  int ng = t * 16 + col, b = ng / CHW, n = ng % CHW;
  int y = n / CW, x = n % CW;
  v8f acc = {};
  #pragma unroll
  for (int k0 = 0; k0 < K3P; k0 += 32) {
    v16h a = load_a16(W3, K3P, k0, lane);
    int kk = k0 + h * 16, tap = kk >> 4;
    v16h bv = {};
    if (tap < 9) {
      int yy = y + tap / 3 - 1, xx = x + tap % 3 - 1;
      if ((unsigned)yy < (unsigned)CH && (unsigned)xx < (unsigned)CW)
        bv = *(const v16h*)(inT + (size_t)(b * CHW + yy * CW + xx) * CI);
    }
    acc = wmma16(a, bv, acc);
  }
  #pragma unroll
  for (int r = 0; r < 8; ++r) { int c = h * 8 + r; out[(size_t)(b * CI + c) * CHW + n] = acc[r]; }
}

// ---------------- 16. BN+relu into concat buffer [pos][32] half ----------------
__global__ void k_bnapply_cc(const float* __restrict__ xin, const float* st,
                             const float* g, const float* bb,
                             _Float16* __restrict__ ccT, int off) {
  int idx = blockIdx.x * 256 + threadIdx.x;
  if (idx >= CB * CI * CHW) return;
  int n = idx % CHW; int c = (idx / CHW) % CI; int b = idx / (CHW * CI);
  float v = (xin[idx] - st[c * 2]) * st[c * 2 + 1] * g[c] + bb[c];
  ccT[(size_t)(b * CHW + n) * 32 + off + c] = (_Float16)fmaxf(v, 0.f);
}

// ---------------- 17. final 1x1: [W6|W6] @ [pa;ca] + bias, relu (single WMMA) ----------------
__global__ void k_gemm_final(const _Float16* __restrict__ ccT, const _Float16* __restrict__ W6,
                             const float* __restrict__ c6b, float* __restrict__ out) {
  int lane = threadIdx.x, t = blockIdx.x, oc0 = blockIdx.y * 16;
  int col = lane & 15, h = lane >> 4;
  int ng = t * 16 + col, b = ng / CHW, n = ng % CHW;
  v16h a = load_a16(W6 + oc0 * 32, 32, 0, lane);
  v16h bv = *(const v16h*)(ccT + (size_t)ng * 32 + h * 16);
  v8f acc = {};
  acc = wmma16(a, bv, acc);
  #pragma unroll
  for (int r = 0; r < 8; ++r) {
    int oc = oc0 + h * 8 + r;
    out[(size_t)(b * CC + oc) * CHW + n] = fmaxf(acc[r] + c6b[oc], 0.f);
  }
}

// ---------------- host ----------------
extern "C" void kernel_launch(void* const* d_in, const int* in_sizes, int n_in,
                              void* d_out, int out_size, void* d_ws, size_t ws_size,
                              hipStream_t stream) {
  (void)in_sizes; (void)n_in; (void)out_size; (void)ws_size;
  const float* z      = (const float*)d_in[0];
  const float* w_at0  = (const float*)d_in[1];  const float* b_at0 = (const float*)d_in[2];
  const float* w_at1  = (const float*)d_in[3];  const float* b_at1 = (const float*)d_in[4];
  const float* w_at2  = (const float*)d_in[5];  const float* b_at2 = (const float*)d_in[6];
  const float* w_at3  = (const float*)d_in[7];  const float* b_at3 = (const float*)d_in[8];
  const float* gn_w   = (const float*)d_in[9];  const float* gn_b  = (const float*)d_in[10];
  const float* conv_w = (const float*)d_in[11]; const float* conv_b= (const float*)d_in[12];
  const float* c5a_w  = (const float*)d_in[13]; const float* bn5a_w= (const float*)d_in[14]; const float* bn5a_b=(const float*)d_in[15];
  const float* c5c_w  = (const float*)d_in[16]; const float* bn5c_w= (const float*)d_in[17]; const float* bn5c_b=(const float*)d_in[18];
  const float* pq_w   = (const float*)d_in[19]; const float* pq_b  = (const float*)d_in[20];
  const float* pk_w   = (const float*)d_in[21]; const float* pk_b  = (const float*)d_in[22];
  const float* pv_w   = (const float*)d_in[23]; const float* pv_b  = (const float*)d_in[24];
  const float* pa_g   = (const float*)d_in[25]; const float* ca_g  = (const float*)d_in[26];
  const float* c5b_w  = (const float*)d_in[27]; const float* bn5b_w= (const float*)d_in[28]; const float* bn5b_b=(const float*)d_in[29];
  const float* c5d_w  = (const float*)d_in[30]; const float* bn5d_w= (const float*)d_in[31]; const float* bn5d_b=(const float*)d_in[32];
  const float* c6_w   = (const float*)d_in[33]; const float* c6_b  = (const float*)d_in[34];
  float* out = (float*)d_out;

  char* wsb = (char*)d_ws;
  size_t off = 0;
  auto alloc = [&](size_t bytes) -> void* {
    void* p = wsb + off;
    off = (off + bytes + 255) & ~(size_t)255;
    return p;
  };
  float*    zc     = (float*)   alloc((size_t)CB * CC * CHW * 4);
  _Float16* xT     = (_Float16*)alloc((size_t)NALL * CC * 2);
  _Float16* zc2T   = (_Float16*)alloc((size_t)NALL * CC * 2);
  _Float16* W1h    = (_Float16*)alloc(65536 * 2);
  _Float16* W2a    = (_Float16*)alloc(16 * K5 * 2);
  _Float16* W2c    = (_Float16*)alloc(16 * K5 * 2);
  _Float16* W3b    = (_Float16*)alloc(16 * K3P * 2);
  _Float16* W3d    = (_Float16*)alloc(16 * K3P * 2);
  _Float16* W6     = (_Float16*)alloc(256 * 32 * 2);
  float*    gstats = (float*)   alloc(8 * 2 * 4);
  float*    c5aout = (float*)   alloc((size_t)CB * CI * CHW * 4);
  float*    c5cout = (float*)   alloc((size_t)CB * CI * CHW * 4);
  float*    st5a   = (float*)   alloc(16 * 2 * 4);
  float*    st5c   = (float*)   alloc(16 * 2 * 4);
  float*    st5b   = (float*)   alloc(16 * 2 * 4);
  float*    st5d   = (float*)   alloc(16 * 2 * 4);
  _Float16* feat1h = (_Float16*)alloc((size_t)CB * CI * CHW * 2);
  _Float16* feat2h = (_Float16*)alloc((size_t)CB * CI * CHW * 2);
  float*    qb     = (float*)   alloc((size_t)CB * 2 * CHW * 4);
  float*    kb     = (float*)   alloc((size_t)CB * 2 * CHW * 4);
  _Float16* vh     = (_Float16*)alloc((size_t)CB * CI * CHW * 2);
  float*    rmax   = (float*)   alloc((size_t)NALL * 4);
  float*    rsum   = (float*)   alloc((size_t)NALL * 4);
  _Float16* paT    = (_Float16*)alloc((size_t)NALL * CI * 2);
  _Float16* caT    = (_Float16*)alloc((size_t)NALL * CI * 2);
  float*    egram  = (float*)   alloc(CB * 256 * 4);
  float*    attc   = (float*)   alloc(CB * 256 * 4);
  float*    c5bout = (float*)   alloc((size_t)CB * CI * CHW * 4);
  float*    c5dout = (float*)   alloc((size_t)CB * CI * CHW * 4);
  _Float16* ccT    = (_Float16*)alloc((size_t)NALL * 32 * 2);

  const int ELTS = CB * CC * CHW;                       // 3,276,800
  k_depthwise<<<(ELTS + 255) / 256, 256, 0, stream>>>(z, w_at0, b_at0, w_at1, b_at1, w_at2, b_at2, w_at3, b_at3, zc);
  k_gnstats<<<8, 256, 0, stream>>>(zc, gstats);
  k_gnapply<<<(ELTS + 255) / 256, 256, 0, stream>>>(zc, gstats, gn_w, gn_b, xT);
  k_prep_weights<<<(152576 + 255) / 256, 256, 0, stream>>>(conv_w, c5a_w, c5c_w, c5b_w, c5d_w, c6_w,
                                                           W1h, W2a, W2c, W3b, W3d, W6);
  k_gemm_pw1<<<dim3(NALL / 16, CC / 16), 32, 0, stream>>>(xT, W1h, conv_b, zc2T);

  dim3 blk4(32, 4);
  k_gemm_c5<<<NALL / 64, blk4, 0, stream>>>(W2a, zc2T, c5aout);
  k_gemm_c5<<<NALL / 64, blk4, 0, stream>>>(W2c, zc2T, c5cout);
  k_bnstats16<<<16, 256, 0, stream>>>(c5aout, st5a);
  k_bnstats16<<<16, 256, 0, stream>>>(c5cout, st5c);
  const int E16 = CB * CI * CHW;                        // 204,800
  k_bnapply16<<<(E16 + 255) / 256, 256, 0, stream>>>(c5aout, st5a, bn5a_w, bn5a_b, feat1h);
  k_bnapply16<<<(E16 + 255) / 256, 256, 0, stream>>>(c5cout, st5c, bn5c_w, bn5c_b, feat2h);

  k_qkv<<<(NALL + 255) / 256, 256, 0, stream>>>(feat1h, pq_w, pq_b, pk_w, pk_b, pv_w, pv_b, qb, kb, vh);
  k_pam_rowstats<<<(NALL + 255) / 256, 256, 0, stream>>>(qb, kb, rmax, rsum);
  k_pam_pv<<<NALL / 64, blk4, 0, stream>>>(qb, kb, rmax, rsum, vh, feat1h, pa_g, paT);

  k_cam_gram<<<CB, 32, 0, stream>>>(feat2h, egram);
  k_cam_softmax<<<CB, 32, 0, stream>>>(egram, attc);
  k_cam_apply<<<(NALL + 255) / 256, 256, 0, stream>>>(feat2h, attc, ca_g, caT);

  k_gemm_small<<<NALL / 64, blk4, 0, stream>>>(W3b, paT, c5bout);
  k_gemm_small<<<NALL / 64, blk4, 0, stream>>>(W3d, caT, c5dout);
  k_bnstats16<<<16, 256, 0, stream>>>(c5bout, st5b);
  k_bnstats16<<<16, 256, 0, stream>>>(c5dout, st5d);
  k_bnapply_cc<<<(E16 + 255) / 256, 256, 0, stream>>>(c5bout, st5b, bn5b_w, bn5b_b, ccT, 0);
  k_bnapply_cc<<<(E16 + 255) / 256, 256, 0, stream>>>(c5dout, st5d, bn5d_w, bn5d_b, ccT, 16);

  k_gemm_final<<<dim3(NALL / 16, CC / 16), 32, 0, stream>>>(ccT, W6, c6_b, out);
}
